// PermuterBlock_53008486367565
// MI455X (gfx1250) — matrix operand order repaired and checked by reading, not compiled, not run on hardware
//
#include <hip/hip_runtime.h>
#include <hip/hip_bf16.h>

// ---------------------------------------------------------------------------
// MI455X (gfx1250) PermuterBlock.
// - All dense GEMMs: v_wmma_f32_16x16x32_f16 (f16 in, f32 accumulate).
// - Weights pre-transposed (N x K); every WMMA operand fragment but bucket-mix A is
//   a ds_load_b128 pair (no half-packing movs in hot loops).
// - Tiles staged global->LDS by the Tensor Data Mover (tensor_load_to_lds),
//   double-buffered in the GEMM (s_wait_tensorcnt 2/0 pipeline).
// Shapes fixed by reference: B=4, N=4096, D=1024, H=8, BS=64, DH=128,
// nb=64, BH=32, M=B*N=16384.
// ---------------------------------------------------------------------------

typedef __attribute__((ext_vector_type(16))) _Float16 v16h;
typedef __attribute__((ext_vector_type(8)))  _Float16 v8h;
typedef __attribute__((ext_vector_type(8)))  float    v8f;
typedef __attribute__((ext_vector_type(4)))  uint32_t v4u;
typedef __attribute__((ext_vector_type(8)))  uint32_t v8u;

union HFrag { v16h v; v8h h[2]; uint32_t u[8]; _Float16 e[16]; };

__device__ __forceinline__ v8f wmma16(v16h a, v16h b, v8f c) {
  return __builtin_amdgcn_wmma_f32_16x16x32_f16(false, a, false, b, (short)0, c,
                                                false, false);
}

// ---------------------------------------------------------------------------
// Tensor Data Mover: 2D tile (f16) global -> LDS, wave-uniform D# in SGPRs.
// tile_w/tile_h/stride in 2-byte elements. pad codes per ISA: interval
// 4 => every 32 DW, 5 => every 64 DW; amount 3 => 4 DW, 35 => 36 DW.
// ---------------------------------------------------------------------------
__device__ __forceinline__ void tdm_load_2d(uint32_t lds_off, const void* gptr,
                                            uint32_t tile_w, uint32_t tile_h,
                                            uint32_t stride, int pad_en,
                                            uint32_t pad_int_code,
                                            uint32_t pad_amt_code) {
  const uint64_t ga = (uint64_t)(uintptr_t)gptr;
  v4u g0;
  g0[0] = 1u;                                   // count=1, user mode
  g0[1] = lds_off;                              // lds_addr (bytes)
  g0[2] = (uint32_t)ga;                         // global_addr[31:0]
  g0[3] = ((uint32_t)(ga >> 32) & 0x01FFFFFFu)  // global_addr[56:32]
          | (2u << 30);                         // type = 2 (image)
  v8u g1;
  uint32_t w0 = (1u << 16);                     // data_size = 2 bytes
  if (pad_en) w0 |= (1u << 20) | (pad_int_code << 22) | (pad_amt_code << 25);
  g1[0] = w0;
  g1[1] = (tile_w & 0xFFFFu) << 16;                        // tensor_dim0[15:0]
  g1[2] = (tile_w >> 16) | ((tile_h & 0xFFFFu) << 16);
  g1[3] = (tile_h >> 16) | ((tile_w & 0xFFFFu) << 16);     // tile_dim0
  g1[4] = tile_h & 0xFFFFu;                                // tile_dim1
  g1[5] = stride;                                          // dim0_stride lo
  g1[6] = 0u;
  g1[7] = 0u;
  asm volatile("tensor_load_to_lds %0, %1" :: "s"(g0), "s"(g1) : "memory");
}

// 16 f16 from LDS as two 16B reads (ds_load_b128 pair),
// second chunk `gap` halves after the first.
__device__ __forceinline__ v16h ld2x8(const _Float16* p, int gap) {
  HFrag f;
  f.h[0] = *(const v8h*)p;
  f.h[1] = *(const v8h*)(p + gap);
  return f.v;
}

// A fragment straight from global memory (row-major, pairs -> dword loads).
__device__ __forceinline__ v16h load_a16_g(const _Float16* A, int lda, int row0,
                                           int k0, int lane) {
  const int hi = (lane >> 4) & 1;
  const int row = row0 + (lane & 15);
  const _Float16* p = A + (size_t)row * lda + k0 + hi * 8;
  HFrag f;
#pragma unroll
  for (int v = 0; v < 8; ++v)
    f.u[v] = *(const uint32_t*)(p + 2 * v + ((v >= 4) ? 8 : 0));
  return f.v;
}

__device__ __forceinline__ float gelu_f(float x) {
  return 0.5f * x * (1.0f + erff(x * 0.70710678118654752440f));
}

// ------------------------------ elementwise --------------------------------

// Convert f32 weight (K x N) to f16 transposed (N x K), tiled through LDS.
__global__ void __launch_bounds__(256) f32_to_f16_t_kernel(
    const float* __restrict__ src, _Float16* __restrict__ dst, int K, int N) {
  __shared__ float tile[32][33];
  const int kb = blockIdx.y * 32, nb = blockIdx.x * 32;
  const int tx = threadIdx.x & 31, ty = threadIdx.x >> 5;
#pragma unroll
  for (int i = ty; i < 32; i += 8)
    tile[i][tx] = src[(size_t)(kb + i) * N + nb + tx];
  __syncthreads();
#pragma unroll
  for (int i = ty; i < 32; i += 8)
    dst[(size_t)(nb + i) * K + kb + tx] = (_Float16)tile[tx][i];
}

__global__ void __launch_bounds__(256) mul_add_kernel(
    const float* __restrict__ u, const float* __restrict__ v,
    const float* __restrict__ x, float* __restrict__ o, int n) {
  int i = blockIdx.x * 256 + threadIdx.x;
  if (i < n) o[i] = u[i] * v[i] + x[i];
}

__global__ void __launch_bounds__(256) layernorm_kernel(
    const float* __restrict__ x, const float* __restrict__ g,
    const float* __restrict__ b, float* __restrict__ y32,
    _Float16* __restrict__ y16) {
  const int row = blockIdx.x;
  const int tid = threadIdx.x;
  const float* xr = x + (size_t)row * 1024;
  __shared__ float red[256];
  float v[4];
  float s = 0.f;
#pragma unroll
  for (int i = 0; i < 4; ++i) { v[i] = xr[tid + i * 256]; s += v[i]; }
  red[tid] = s; __syncthreads();
  for (int o = 128; o > 0; o >>= 1) {
    if (tid < o) red[tid] += red[tid + o];
    __syncthreads();
  }
  const float mu = red[0] * (1.0f / 1024.0f);
  __syncthreads();
  float s2 = 0.f;
#pragma unroll
  for (int i = 0; i < 4; ++i) { float d = v[i] - mu; s2 += d * d; }
  red[tid] = s2; __syncthreads();
  for (int o = 128; o > 0; o >>= 1) {
    if (tid < o) red[tid] += red[tid + o];
    __syncthreads();
  }
  const float inv = rsqrtf(red[0] * (1.0f / 1024.0f) + 1e-5f);
#pragma unroll
  for (int i = 0; i < 4; ++i) {
    const int c = tid + i * 256;
    const float o = (v[i] - mu) * inv * g[c] + b[c];
    if (y32) y32[(size_t)row * 1024 + c] = o;
    if (y16) y16[(size_t)row * 1024 + c] = (_Float16)o;
  }
}

// --------------------------- generic WMMA GEMM -----------------------------
// C[M,N] = A[M,K] @ Wt[N,K]^T. Block = 8 waves, 128(M) x 128(N); K staged
// 128 at a time by double-buffered TDM (LDS stride 136 halves). Wave tile
// 32x64 (2x4 wmma): 12 ds_load_b128 per 8 wmma. FVT: also emit a per-head
// transposed copy of the V half (cols >= 1024) as vT[b,h,dh,n].

enum { FB = 1, FR = 2, FG = 4, F32 = 8, F16 = 16, FVT = 32 };

template <int FLAGS>
__global__ void __launch_bounds__(256) gemm_wmma(
    const _Float16* __restrict__ A, const _Float16* __restrict__ Wt,
    const float* __restrict__ bias, const float* __restrict__ resid,
    float* __restrict__ out32, _Float16* __restrict__ out16,
    _Float16* __restrict__ vt, int M, int N, int K) {
  __shared__ _Float16 As[2][128 * 136];
  __shared__ _Float16 Bs[2][128 * 136];
  const int lane = threadIdx.x & 31;
  const int wave = threadIdx.x >> 5;
  const int mb = (wave & 3) * 32;
  const int nbw = (wave >> 2) * 64;
  const int hi = (lane >> 4) & 1;
  const int hi8 = hi * 8, hi16 = hi * 16, cc = lane & 15;
  const size_t gm0 = (size_t)blockIdx.x * 128;
  const size_t gn0 = (size_t)blockIdx.y * 128;
  (void)M;
  const int niter = K >> 7;
  v8f acc[2][4] = {};

  auto issue = [&](int it) {
    const int kt = it << 7;
    // rows of 64 DWORDs + 4 DWORD pad -> LDS stride 136 halves
    tdm_load_2d((uint32_t)(uintptr_t)&As[it & 1][0], A + gm0 * K + kt, 128, 128,
                (uint32_t)K, 1, 5, 3);
    tdm_load_2d((uint32_t)(uintptr_t)&Bs[it & 1][0], Wt + gn0 * K + kt, 128,
                128, (uint32_t)K, 1, 5, 3);
  };
  if (threadIdx.x < 32) issue(0);
  for (int it = 0; it < niter; ++it) {
    if (it + 1 < niter) {
      if (threadIdx.x < 32) issue(it + 1);
      __builtin_amdgcn_s_wait_tensorcnt(2);  // tile `it` complete (in-order)
    } else {
      __builtin_amdgcn_s_wait_tensorcnt(0);
    }
    __syncthreads();
    const _Float16* Ab = &As[it & 1][0];
    const _Float16* Bb = &Bs[it & 1][0];
#pragma unroll
    for (int k0 = 0; k0 < 128; k0 += 32) {
      v16h a0 = ld2x8(&Ab[(mb + cc) * 136 + k0 + hi8], 16);
      v16h a1 = ld2x8(&Ab[(mb + 16 + cc) * 136 + k0 + hi8], 16);
#pragma unroll
      for (int j = 0; j < 4; ++j) {
        v16h bj = ld2x8(&Bb[(nbw + j * 16 + cc) * 136 + k0 + hi8], 16);
        acc[0][j] = wmma16(a0, bj, acc[0][j]);
        acc[1][j] = wmma16(a1, bj, acc[1][j]);
      }
    }
    __syncthreads();  // compute done before buffer refilled
  }
  const int hi8e = hi * 8;
#pragma unroll
  for (int mi = 0; mi < 2; ++mi)
#pragma unroll
    for (int j = 0; j < 4; ++j) {
      v8f a = acc[mi][j];
      const int col = (int)gn0 + nbw + j * 16 + cc;
      const float bv = (FLAGS & FB) ? bias[col] : 0.0f;
#pragma unroll
      for (int v = 0; v < 8; ++v) {
        const int row = (int)gm0 + mb + mi * 16 + v + hi8e;
        float x = a[v] + bv;
        if (FLAGS & FR) x += resid[(size_t)row * N + col];
        if (FLAGS & FG) x = gelu_f(x);
        if (FLAGS & F32) out32[(size_t)row * N + col] = x;
        if (FLAGS & F16) out16[(size_t)row * N + col] = (_Float16)x;
        if (FLAGS & FVT) {
          if (col >= 1024) {  // V half -> vT[b, h, dh, n]
            const int hh = (col - 1024) >> 7, dh = (col - 1024) & 127;
            const int bb = row >> 12, nn = row & 4095;
            vt[(((size_t)bb * 8 + hh) * 128 + dh) * 4096 + nn] = (_Float16)x;
          }
        }
      }
    }
}

// ----------------------------- bucket means --------------------------------
__global__ void __launch_bounds__(128) bucket_mean_kernel(
    const _Float16* __restrict__ src, int ld, _Float16* __restrict__ dst) {
  const int u = blockIdx.x, bh = blockIdx.y, d = threadIdx.x;
  const int b = bh >> 3, h = bh & 7;
  const _Float16* p = src + ((size_t)(b * 4096 + u * 64) * ld + h * 128 + d);
  float s = 0.f;
#pragma unroll 8
  for (int t = 0; t < 64; ++t) s += (float)p[(size_t)t * ld];
  dst[((size_t)bh * 64 + u) * 128 + d] = (_Float16)(s * (1.0f / 64.0f));
}

// ------------------------------- sinkhorn ----------------------------------
__global__ void __launch_bounds__(64) sinkhorn_kernel(
    const _Float16* __restrict__ sq, const _Float16* __restrict__ sk,
    _Float16* __restrict__ R) {
  __shared__ _Float16 sqs[64 * 128];
  __shared__ _Float16 sks[64 * 128];
  __shared__ float r[64][65];
  const int bh = blockIdx.x, t = threadIdx.x;
  for (int i = 0; i < 128; ++i) {
    sqs[t * 128 + i] = sq[((size_t)bh * 64 + t) * 128 + i];
    sks[t * 128 + i] = sk[((size_t)bh * 64 + t) * 128 + i];
  }
  __syncthreads();
  const float scale = 0.08838834764831845f;  // 128^-0.5
  for (int i = 0; i < 64; ++i) {
    float s = 0.f;
    for (int d = 0; d < 128; ++d)
      s += (float)sqs[i * 128 + d] * (float)sks[t * 128 + d];
    s = fmaxf(s * scale, 0.0f) + 1e-6f;
    r[i][t] = logf(s) * (1.0f / 0.75f);
  }
  __syncthreads();
  for (int it = 0; it < 8; ++it) {
    float m = -3.0e38f;
    for (int j = 0; j < 64; ++j) m = fmaxf(m, r[t][j]);
    float s = 0.f;
    for (int j = 0; j < 64; ++j) s += expf(r[t][j] - m);
    float lse = m + logf(s);
    for (int j = 0; j < 64; ++j) r[t][j] -= lse;
    __syncthreads();
    m = -3.0e38f;
    for (int i = 0; i < 64; ++i) m = fmaxf(m, r[i][t]);
    s = 0.f;
    for (int i = 0; i < 64; ++i) s += expf(r[i][t] - m);
    lse = m + logf(s);
    for (int i = 0; i < 64; ++i) r[i][t] -= lse;
    __syncthreads();
  }
  for (int i = 0; i < 64; ++i)
    R[((size_t)bh * 64 + i) * 64 + t] = (_Float16)expf(r[i][t]);
}

// ------------------------------ bucket mix ---------------------------------
// out = R @ bucket-matrix. Each block: one bh, 64 output cols (fixed t).
// The 64(vb)x64(d) slab is staged transposed into LDS so B fragments are
// ds_load_b128 pairs. TR=1 stores output transposed: [bh][u][d][t].
template <int TR>
__global__ void __launch_bounds__(256) bucket_mix_kernel(
    const _Float16* __restrict__ R, const _Float16* __restrict__ kv, int voff,
    _Float16* __restrict__ out) {
  __shared__ _Float16 BsT[64 * 72];  // [d 0..63][vb 0..63], stride 72
  const int tid = threadIdx.x;
  const int lane = tid & 31;
  const int wave = tid >> 5;
  const int bh = blockIdx.y;
  const int b = bh >> 3, h = bh & 7;
  const int bx = blockIdx.x;         // 64-col group: t = bx>>1 fixed
  const int tt = bx >> 1;
  const int dbase = (bx & 1) * 64;
  // stage + transpose the slab (coalesced b128 reads, b16 scattered stores)
  {
    const _Float16* src =
        kv + ((size_t)(b * 4096 + tt)) * 2048 + h * 128 + dbase + voff;
#pragma unroll
    for (int ci = 0; ci < 2; ++ci) {
      const int c = tid * 2 + ci;       // 512 chunks of 8 halves
      const int vb = c >> 3, doff = (c & 7) * 8;
      union { v8h v; _Float16 e[8]; } tmp;
      tmp.v = *(const v8h*)(src + (size_t)vb * (64 * 2048) + doff);
#pragma unroll
      for (int e = 0; e < 8; ++e) BsT[(doff + e) * 72 + vb] = tmp.e[e];
    }
  }
  __syncthreads();
  const int wm = wave & 3;
  const int ncol = (wave >> 2) * 32;
  const int hi = (lane >> 4) & 1, hi16 = hi * 16, cc = lane & 15;
  const _Float16* Rb = R + (size_t)bh * 64 * 64;
  v8f acc0 = {}, acc1 = {};
#pragma unroll
  for (int k0 = 0; k0 < 64; k0 += 32) {
    v16h a = load_a16_g(Rb, 64, wm * 16, k0, lane);
    const int kb = k0 + hi * 8;
    v16h b0 = ld2x8(&BsT[(ncol + cc) * 72 + kb], 16);
    v16h b1 = ld2x8(&BsT[(ncol + 16 + cc) * 72 + kb], 16);
    acc0 = wmma16(a, b0, acc0);
    acc1 = wmma16(a, b1, acc1);
  }
  const int hi8 = hi * 8;
#pragma unroll
  for (int nt = 0; nt < 2; ++nt) {
    v8f a = nt ? acc1 : acc0;
    const int l = ncol + nt * 16 + cc;  // local col (d within half)
#pragma unroll
    for (int v = 0; v < 8; ++v) {
      const int u = wm * 16 + v + hi8;
      const size_t base = ((size_t)bh * 64 + u) * 8192;
      if (TR) out[base + (size_t)(dbase + l) * 64 + tt] = (_Float16)a[v];
      else    out[base + (size_t)tt * 128 + dbase + l] = (_Float16)a[v];
    }
  }
}

// ------------------------------- attention ---------------------------------
// Per (bucket u, batch-head bh): q(64x128) vs 128 keys (64 routed + 64 local).
// q/k2 row-major, v2 transposed [dh][key]; all staged by five TDM
// descriptors. Every fragment is a ds_load_b128 pair.
__global__ void __launch_bounds__(256) attention_kernel(
    const _Float16* __restrict__ q16, const _Float16* __restrict__ kv16,
    const _Float16* __restrict__ bkr, const _Float16* __restrict__ bvrT,
    const _Float16* __restrict__ vT, _Float16* __restrict__ ao) {
  __shared__ _Float16 qs[64 * 136];
  __shared__ _Float16 ks[128 * 136];
  __shared__ _Float16 vsT[128 * 136];  // [dh][key]
  __shared__ float dots[64 * 128];
  __shared__ _Float16 attn[64 * 128];
  const int u = blockIdx.x, bh = blockIdx.y;
  const int b = bh >> 3, h = bh & 7;
  const int tid = threadIdx.x, lane = tid & 31, wave = tid >> 5;
  const int wm = wave & 3;
  const int nb0 = (wave >> 2) * 64;
  const int hi = (lane >> 4) & 1, hi8 = hi * 8, hi16 = hi * 16, cc = lane & 15;
  const size_t qrow0 = (size_t)(b * 4096 + u * 64);
  const size_t rbase = ((size_t)bh * 64 + u) * 8192;

  if (tid < 32) {
    // 128-wide rows: 64 DW + 4 DW pad -> stride 136
    tdm_load_2d((uint32_t)(uintptr_t)qs, q16 + qrow0 * 1024 + h * 128,
                128, 64, 1024, 1, 5, 3);
    tdm_load_2d((uint32_t)(uintptr_t)ks, bkr + rbase, 128, 64, 128, 1, 5, 3);
    tdm_load_2d((uint32_t)(uintptr_t)(ks + 64 * 136),
                kv16 + qrow0 * 2048 + h * 128, 128, 64, 2048, 1, 5, 3);
    // 64-wide rows into 136-stride: 32 DW + 36 DW pad; two tiles interleave
    // keys [0,64) (routed, bvrT[d][t]) and [64,128) (local, vT[b,h,dh,n]).
    tdm_load_2d((uint32_t)(uintptr_t)vsT, bvrT + rbase, 64, 128, 64, 1, 4, 35);
    tdm_load_2d((uint32_t)(uintptr_t)(vsT + 64),
                vT + ((size_t)(b * 8 + h) * 128) * 4096 + u * 64,
                64, 128, 4096, 1, 4, 35);
  }
  __builtin_amdgcn_s_wait_tensorcnt(0);
  __syncthreads();

  // Phase 1: dots = q @ k2^T  (M=64 rows, N=128 keys, K=128 dh).
  v8f acc[4] = {};
#pragma unroll
  for (int k0 = 0; k0 < 128; k0 += 32) {
    v16h a = ld2x8(&qs[(wm * 16 + cc) * 136 + k0 + hi8], 16);
#pragma unroll
    for (int j = 0; j < 4; ++j) {
      const int key = nb0 + j * 16 + cc;
      v16h bf = ld2x8(&ks[key * 136 + k0 + hi8], 16);
      acc[j] = wmma16(a, bf, acc[j]);
    }
  }
  const float scale = 0.08838834764831845f;
  const int hi8e = hi * 8;
#pragma unroll
  for (int j = 0; j < 4; ++j)
#pragma unroll
    for (int v = 0; v < 8; ++v)
      dots[(wm * 16 + v + hi8e) * 128 + nb0 + j * 16 + cc] = acc[j][v] * scale;
  __syncthreads();

  // Phase 2: row softmax
  if (tid < 64) {
    float m = -3.0e38f;
    for (int j = 0; j < 128; ++j) m = fmaxf(m, dots[tid * 128 + j]);
    float s = 0.f;
    for (int j = 0; j < 128; ++j) s += expf(dots[tid * 128 + j] - m);
    const float inv = 1.0f / s;
    for (int j = 0; j < 128; ++j)
      attn[tid * 128 + j] = (_Float16)(expf(dots[tid * 128 + j] - m) * inv);
  }
  __syncthreads();

  // Phase 3: out = attn @ v2  (M=64, N=128 dh, K=128 keys), B from vsT.
  v8f acc2[4] = {};
#pragma unroll
  for (int k0 = 0; k0 < 128; k0 += 32) {
    v16h a = ld2x8(&attn[(wm * 16 + cc) * 128 + k0 + hi8], 16);
    const int kb = k0 + hi8;
#pragma unroll
    for (int j = 0; j < 4; ++j) {
      const int dh = nb0 + j * 16 + cc;
      v16h bf = ld2x8(&vsT[dh * 136 + kb], 16);
      acc2[j] = wmma16(a, bf, acc2[j]);
    }
  }
#pragma unroll
  for (int j = 0; j < 4; ++j)
#pragma unroll
    for (int v = 0; v < 8; ++v) {
      const int t = wm * 16 + v + hi8e;
      const int dh = nb0 + j * 16 + cc;
      ao[(qrow0 + t) * 1024 + h * 128 + dh] = (_Float16)acc2[j][v];
    }
}

// ------------------------------- launcher ----------------------------------

extern "C" void kernel_launch(void* const* d_in, const int* in_sizes, int n_in,
                              void* d_out, int out_size, void* d_ws,
                              size_t ws_size, hipStream_t stream) {
  (void)in_sizes; (void)n_in; (void)out_size; (void)ws_size;
  const float* x      = (const float*)d_in[0];
  const float* ln_g   = (const float*)d_in[1];
  const float* ln_b   = (const float*)d_in[2];
  const float* proj_w = (const float*)d_in[3];
  const float* proj_b = (const float*)d_in[4];
  const float* aln_g  = (const float*)d_in[5];
  const float* aln_b  = (const float*)d_in[6];
  const float* wq     = (const float*)d_in[7];
  const float* wkv    = (const float*)d_in[8];
  const float* wo     = (const float*)d_in[9];
  const float* bo     = (const float*)d_in[10];
  const float* fln_g  = (const float*)d_in[11];
  const float* fln_b  = (const float*)d_in[12];
  const float* pw1    = (const float*)d_in[13];
  const float* pb1    = (const float*)d_in[14];
  const float* pw2    = (const float*)d_in[15];
  const float* pb2    = (const float*)d_in[16];
  const float* fw1    = (const float*)d_in[17];
  const float* fb1    = (const float*)d_in[18];
  const float* fw2    = (const float*)d_in[19];
  const float* fb2    = (const float*)d_in[20];
  float* out = (float*)d_out;

  const int M = 16384, D = 1024;
  const size_t MD32  = (size_t)M * D * 4;
  const size_t MD16  = (size_t)M * D * 2;
  const size_t M4D16 = (size_t)M * 4096 * 2;

  char* ws = (char*)d_ws;
  size_t o = 0;
  auto alloc = [&](size_t bytes) {
    size_t r = o; o += (bytes + 255) & ~(size_t)255; return r;
  };
  _Float16* proj_wT = (_Float16*)(ws + alloc((size_t)D * D * 2));
  _Float16* wqT     = (_Float16*)(ws + alloc((size_t)D * D * 2));
  _Float16* wkvT    = (_Float16*)(ws + alloc((size_t)D * 2 * D * 2));
  _Float16* woT     = (_Float16*)(ws + alloc((size_t)D * D * 2));
  _Float16* pw1T    = (_Float16*)(ws + alloc((size_t)D * 4096 * 2));
  _Float16* pw2T    = (_Float16*)(ws + alloc((size_t)4096 * D * 2));
  _Float16* fw1T    = (_Float16*)(ws + alloc((size_t)D * 4096 * 2));
  _Float16* fw2T    = (_Float16*)(ws + alloc((size_t)4096 * D * 2));
  float*    h32   = (float*)(ws + alloc(MD32));       // later xm32
  float*    u32   = (float*)(ws + alloc(MD32));
  _Float16* kv16  = (_Float16*)(ws + alloc(MD32));    // M x 2048 f16
  _Float16* vT16  = (_Float16*)(ws + alloc(MD16));    // [b,h,dh,n]
  _Float16* q16   = (_Float16*)(ws + alloc(MD16));
  _Float16* h16   = (_Float16*)(ws + alloc(MD16));    // later t2_16
  _Float16* h2_16 = (_Float16*)(ws + alloc(MD16));    // later t16
  float*    x2_32 = (float*)(ws + alloc(MD32));
  _Float16* sq16  = (_Float16*)(ws + alloc((size_t)32 * 64 * 128 * 2));
  _Float16* sk16  = (_Float16*)(ws + alloc((size_t)32 * 64 * 128 * 2));
  _Float16* R16   = (_Float16*)(ws + alloc((size_t)32 * 64 * 64 * 2));
  char*     big   = ws + alloc(M4D16);                // bkr+bvrT, later g16
  _Float16* bkr16  = (_Float16*)big;
  _Float16* bvrT16 = (_Float16*)(big + (size_t)32 * 64 * 64 * 128 * 2);
  _Float16* g16    = (_Float16*)big;
  _Float16* ao16  = (_Float16*)(ws + alloc(MD16));
  float*    v32   = (float*)(ws + alloc(MD32));
  float*    xm32  = h32;
  _Float16* t16   = h2_16;
  _Float16* t2_16 = h16;

  auto cvtT = [&](const float* s, _Float16* d, int K, int N) {
    f32_to_f16_t_kernel<<<dim3(N / 32, K / 32), 256, 0, stream>>>(s, d, K, N);
  };
  cvtT(proj_w, proj_wT, D, D);
  cvtT(wq, wqT, D, D);
  cvtT(wkv, wkvT, D, 2 * D);
  cvtT(wo, woT, D, D);
  cvtT(pw1, pw1T, D, 4096);
  cvtT(pw2, pw2T, 4096, D);
  cvtT(fw1, fw1T, D, 4096);
  cvtT(fw2, fw2T, 4096, D);

  dim3 blk(256);
  layernorm_kernel<<<M, blk, 0, stream>>>(x, ln_g, ln_b, h32, h16);
  gemm_wmma<FB | F32><<<dim3(M / 128, D / 128), blk, 0, stream>>>(
      h16, proj_wT, proj_b, nullptr, u32, nullptr, nullptr, M, D, D);
  layernorm_kernel<<<M, blk, 0, stream>>>(h32, aln_g, aln_b, nullptr, h2_16);
  gemm_wmma<F16><<<dim3(M / 128, D / 128), blk, 0, stream>>>(
      h2_16, wqT, nullptr, nullptr, nullptr, q16, nullptr, M, D, D);
  gemm_wmma<F16 | FVT><<<dim3(M / 128, 2 * D / 128), blk, 0, stream>>>(
      h2_16, wkvT, nullptr, nullptr, nullptr, kv16, vT16, M, 2 * D, D);
  bucket_mean_kernel<<<dim3(64, 32), 128, 0, stream>>>(q16, 1024, sq16);
  bucket_mean_kernel<<<dim3(64, 32), 128, 0, stream>>>(kv16, 2048, sk16);
  sinkhorn_kernel<<<32, 64, 0, stream>>>(sq16, sk16, R16);
  bucket_mix_kernel<0><<<dim3(128, 32), blk, 0, stream>>>(R16, kv16, 0, bkr16);
  bucket_mix_kernel<1><<<dim3(128, 32), blk, 0, stream>>>(R16, kv16, 1024,
                                                          bvrT16);
  attention_kernel<<<dim3(64, 32), blk, 0, stream>>>(q16, kv16, bkr16, bvrT16,
                                                     vT16, ao16);
  gemm_wmma<FB | FR | F32><<<dim3(M / 128, D / 128), blk, 0, stream>>>(
      ao16, woT, bo, h32, x2_32, nullptr, nullptr, M, D, D);
  layernorm_kernel<<<M, blk, 0, stream>>>(x2_32, fln_g, fln_b, nullptr, t16);
  gemm_wmma<FB | FG | F16><<<dim3(M / 128, 4096 / 128), blk, 0, stream>>>(
      t16, pw1T, pb1, nullptr, nullptr, g16, nullptr, M, 4096, D);
  gemm_wmma<FB | FR | F32><<<dim3(M / 128, D / 128), blk, 0, stream>>>(
      g16, pw2T, pb2, x2_32, v32, nullptr, nullptr, M, D, 4096);
  mul_add_kernel<<<(M * D + 255) / 256, blk, 0, stream>>>(u32, v32, x, xm32,
                                                          M * D);
  layernorm_kernel<<<M, blk, 0, stream>>>(xm32, ln_g, ln_b, nullptr, t2_16);
  gemm_wmma<FB | FG | F16><<<dim3(M / 128, 4096 / 128), blk, 0, stream>>>(
      t2_16, fw1T, fb1, nullptr, nullptr, g16, nullptr, M, 4096, D);
  gemm_wmma<FB | FR | F32><<<dim3(M / 128, D / 128), blk, 0, stream>>>(
      g16, fw2T, fb2, xm32, out, nullptr, nullptr, M, D, 4096);
}
